// MultiHeadedAttention_55946243997836
// MI455X (gfx1250) — matrix operand order repaired and checked by reading, not compiled, not run on hardware
//
#include <hip/hip_runtime.h>
#include <hip/hip_bf16.h>

// ---------------------------------------------------------------------------
// Fused multi-head attention block for MI455X (gfx1250, wave32, WMMA).
//   proj = x @ Wp^T + bp ; split q,k,v ; flash-attention ; y = concat @ Wo^T
//   z = x + y + bo ; out = layernorm(z) * gamma + beta
// Matmuls via v_wmma_f32_16x16x32_f16. Attention K/V tiles are double-buffered
// in LDS via global_load_async_to_lds_b128 (ASYNCcnt) so the async engine
// overlaps the WMMA pipe; consumed via ds reads.
// ---------------------------------------------------------------------------

typedef __attribute__((ext_vector_type(16))) _Float16 v16h;
typedef __attribute__((ext_vector_type(8)))  _Float16 v8h;
typedef __attribute__((ext_vector_type(8)))  float    v8f;

#define WMMA_F16(A, B, C) \
  __builtin_amdgcn_wmma_f32_16x16x32_f16(false, (A), false, (B), (short)0, (C), false, false)

static constexpr int BN   = 4;
static constexpr int S    = 2048;
static constexpr int D    = 512;
static constexpr int H    = 6;
static constexpr int DQK  = 64;
static constexpr int DV   = 64;
static constexpr int PROJ = (DQK + DQK + DV) * H;  // 1152
static constexpr int ROWS = BN * S;                // 8192
static constexpr int FEAT = H * DV;                // 384

// workspace layout (bytes, all 256B aligned)
static constexpr size_t OFF_XH     = 0;
static constexpr size_t OFF_WPH    = OFF_XH     + (size_t)ROWS * D * 2;
static constexpr size_t OFF_WOH    = OFF_WPH    + (size_t)PROJ * D * 2;
static constexpr size_t OFF_QH     = OFF_WOH    + (size_t)D * FEAT * 2;
static constexpr size_t OFF_KH     = OFF_QH     + (size_t)BN * H * S * DQK * 2;
static constexpr size_t OFF_VTH    = OFF_KH     + (size_t)BN * H * S * DQK * 2;
static constexpr size_t OFF_CONCAT = OFF_VTH    + (size_t)BN * H * DV * S * 2;
static constexpr size_t OFF_Z      = OFF_CONCAT + (size_t)ROWS * FEAT * 2;

// ---------------------------------------------------------------------------
// CDNA5 async global->LDS copy helpers (per-lane 16B, tracked by ASYNCcnt)
// ---------------------------------------------------------------------------
__device__ __forceinline__ unsigned mha_lds_off(const void* p) {
  // addrspace(3) offsets live in the low 32 bits of the generic address
  return (unsigned)(unsigned long long)p;
}
__device__ __forceinline__ void mha_async_ld_b128(unsigned lds_dst, const void* gsrc) {
  asm volatile("global_load_async_to_lds_b128 %0, %1, off"
               :: "v"(lds_dst), "v"(gsrc) : "memory");
}
__device__ __forceinline__ void mha_wait_async0() {
  asm volatile("s_wait_asynccnt 0x0" ::: "memory");
}
__device__ __forceinline__ void mha_wait_async_le4() {
  // previous chunk's 4 async loads retired; the 4 just issued may be in flight
  asm volatile("s_wait_asynccnt 0x4" ::: "memory");
}
__device__ __forceinline__ void mha_wait_ds0() {
  asm volatile("s_wait_dscnt 0x0" ::: "memory");
}

// ---------------------------------------------------------------------------
// Kernel 0: f32 -> f16 conversion (grid-stride)
// ---------------------------------------------------------------------------
__global__ __launch_bounds__(256)
void mha_cvt_f16(const float* __restrict__ in, _Float16* __restrict__ out, int n) {
  int i = blockIdx.x * blockDim.x + threadIdx.x;
  int stride = gridDim.x * blockDim.x;
  for (; i < n; i += stride) out[i] = (_Float16)in[i];
}

// ---------------------------------------------------------------------------
// Kernel 1: QKV projection. One wave computes a 16(M) x 64(N) tile of proj,
// adds bias and scatters q (pre-scaled by 1/sqrt(DQK)), k, v^T to workspace.
// ---------------------------------------------------------------------------
__global__ __launch_bounds__(32)
void mha_qkv_gemm(const _Float16* __restrict__ xh, const _Float16* __restrict__ Wph,
                  const float* __restrict__ bp,
                  _Float16* __restrict__ qh, _Float16* __restrict__ kh,
                  _Float16* __restrict__ vth) {
  const int lane = threadIdx.x;
  const int n    = lane & 15;
  const int half = lane >> 4;
  const int m0   = blockIdx.y * 16;
  const int p0   = blockIdx.x * 64;

  const v8f vzero = {0.f, 0.f, 0.f, 0.f, 0.f, 0.f, 0.f, 0.f};
  v8f acc[4] = {vzero, vzero, vzero, vzero};

  const _Float16* arow = xh + (size_t)(m0 + n) * D;

  for (int k0 = 0; k0 < D; k0 += 32) {
    __builtin_prefetch(arow + k0 + 64, 0, 0);  // global_prefetch_b8
    union { v16h v; v8h h[2]; } a;
    a.h[0] = *(const v8h*)(arow + k0 + half * 16);
    a.h[1] = *(const v8h*)(arow + k0 + half * 16 + 8);
#pragma unroll
    for (int sub = 0; sub < 4; ++sub) {
      const _Float16* wrow = Wph + (size_t)(p0 + sub * 16 + n) * D;
      v16h b = *(const v16h*)(wrow + k0 + half * 16);
      acc[sub] = WMMA_F16(a.v, b, acc[sub]);
    }
  }

#pragma unroll
  for (int sub = 0; sub < 4; ++sub) {
    const int p    = p0 + sub * 16 + n;
    const float bi = bp[p];
    const int sec  = p / FEAT;        // 0=q 1=k 2=v
    const int rem  = p % FEAT;
    const int hh   = rem >> 6;
    const int dd   = rem & 63;
#pragma unroll
    for (int r = 0; r < 8; ++r) {
      const int row = m0 + r + 8 * half;
      const int bi_ = row >> 11;      // row / S
      const int si  = row & (S - 1);
      const float v = acc[sub][r] + bi;
      const size_t bh = (size_t)(bi_ * H + hh);
      if (sec == 0)      qh[(bh * S + si) * DQK + dd]  = (_Float16)(v * 0.125f);
      else if (sec == 1) kh[(bh * S + si) * DQK + dd]  = (_Float16)v;
      else               vth[(bh * DV + dd) * S + si]  = (_Float16)v;
    }
  }
}

// ---------------------------------------------------------------------------
// Kernel 2: flash attention. 128 threads (4 waves) own 64 query rows of one
// (b,h); each wave computes 16 rows. Per 32-column K/V chunk the block stages
//   Ksh[buf][32][64] (K rows, d-contiguous) and Vsh[buf][64][32] (dv rows,
//   k-contiguous) into LDS via global_load_async_to_lds_b128, double-buffered:
// chunk i+1 streams in while WMMA consumes chunk i (s_wait_asynccnt 4).
// Online softmax per wave; P transposed C/D->A layout through LDS.
// ---------------------------------------------------------------------------
__global__ __launch_bounds__(128)
void mha_flash_attn(const _Float16* __restrict__ qh, const _Float16* __restrict__ kh,
                    const _Float16* __restrict__ vth, _Float16* __restrict__ concat) {
  __shared__ __align__(32) _Float16 Ksh[2][32][DQK];   // 2 x 4 KB
  __shared__ __align__(32) _Float16 Vsh[2][DV][32];    // 2 x 4 KB
  __shared__ __align__(32) _Float16 Pt[4][16][32];     // 4 KB

  const int tid  = threadIdx.x;
  const int wid  = tid >> 5;
  const int lane = tid & 31;
  const int n    = lane & 15;
  const int half = lane >> 4;
  const int bh   = blockIdx.y;                 // 0..B*H-1
  const int q0   = blockIdx.x * 64 + wid * 16; // this wave's query rows
  const size_t base = (size_t)bh * S * DQK;    // same extent for qh/kh/vth

  const _Float16* khb = kh  + base;
  const _Float16* vtb = vth + base;

  // async-copy assignments for this thread (2 K segments + 2 V segments)
  const int ks0 = tid,       ks1 = tid + 128;       // of 256 K segs (16B each)
  const int kr0 = ks0 >> 3,  ko0 = (ks0 & 7) * 8;   // 8 segs per 128B K row
  const int kr1 = ks1 >> 3,  ko1 = (ks1 & 7) * 8;
  const int vs0 = tid,       vs1 = tid + 128;       // of 256 V segs (16B each)
  const int vr0 = vs0 >> 2,  vo0 = (vs0 & 3) * 8;   // 4 segs per 64B V row
  const int vr1 = vs1 >> 2,  vo1 = (vs1 & 3) * 8;
  unsigned ldsK0[2], ldsK1[2], ldsV0[2], ldsV1[2];
#pragma unroll
  for (int b = 0; b < 2; ++b) {
    ldsK0[b] = mha_lds_off(&Ksh[b][kr0][ko0]);
    ldsK1[b] = mha_lds_off(&Ksh[b][kr1][ko1]);
    ldsV0[b] = mha_lds_off(&Vsh[b][vr0][vo0]);
    ldsV1[b] = mha_lds_off(&Vsh[b][vr1][vo1]);
  }

  // Q A-operands for d=0..31 and d=32..63 (held for the whole loop)
  const _Float16* qrow = qh + base + (size_t)(q0 + n) * DQK;
  union { v16h v; v8h h[2]; } aq0, aq1;
  aq0.h[0] = *(const v8h*)(qrow + half * 16);
  aq0.h[1] = *(const v8h*)(qrow + half * 16 + 8);
  aq1.h[0] = *(const v8h*)(qrow + 32 + half * 16);
  aq1.h[1] = *(const v8h*)(qrow + 40 + half * 16);

  const v8f vzero = {0.f, 0.f, 0.f, 0.f, 0.f, 0.f, 0.f, 0.f};
  v8f O[4] = {vzero, vzero, vzero, vzero};
  float run_m[8], run_l[8];
#pragma unroll
  for (int r = 0; r < 8; ++r) { run_m[r] = -3.0e38f; run_l[r] = 0.f; }

  // prologue: stage chunk 0 into buffer 0
  mha_async_ld_b128(ldsK0[0], khb + (size_t)kr0 * DQK + ko0);
  mha_async_ld_b128(ldsK1[0], khb + (size_t)kr1 * DQK + ko1);
  mha_async_ld_b128(ldsV0[0], vtb + (size_t)vr0 * S + vo0);
  mha_async_ld_b128(ldsV1[0], vtb + (size_t)vr1 * S + vo1);

  int cur = 0;
  for (int kc = 0; kc < S; kc += 32) {
    const int nxt = cur ^ 1;
    __syncthreads();  // all waves done reading buf[nxt] (from two chunks ago)

    if (kc + 32 < S) {
      const int kn = kc + 32;
      mha_async_ld_b128(ldsK0[nxt], khb + (size_t)(kn + kr0) * DQK + ko0);
      mha_async_ld_b128(ldsK1[nxt], khb + (size_t)(kn + kr1) * DQK + ko1);
      mha_async_ld_b128(ldsV0[nxt], vtb + (size_t)vr0 * S + kn + vo0);
      mha_async_ld_b128(ldsV1[nxt], vtb + (size_t)vr1 * S + kn + vo1);
      mha_wait_async_le4();  // buf[cur]'s 4 loads retired (in-order completion)
    } else {
      mha_wait_async0();
    }
    __syncthreads();  // buf[cur] visible to all waves

    // scores: two 16x16 tiles, each accumulated over d in two WMMA
    v8f s[2];
#pragma unroll
    for (int j = 0; j < 2; ++j) {
      const _Float16* krow = &Ksh[cur][j * 16 + n][0];
      v16h b0 = *(const v16h*)(krow + half * 16);
      v16h b1 = *(const v16h*)(krow + 32 + half * 16);
      v8f c = vzero;
      c = WMMA_F16(aq0.v, b0, c);
      c = WMMA_F16(aq1.v, b1, c);
      s[j] = c;
    }

    // row-wise max across the 16 k-columns (lanes sharing the same half)
    float t8[8];
#pragma unroll
    for (int r = 0; r < 8; ++r) t8[r] = fmaxf(s[0][r], s[1][r]);
#pragma unroll
    for (int off = 1; off < 16; off <<= 1)
#pragma unroll
      for (int r = 0; r < 8; ++r) t8[r] = fmaxf(t8[r], __shfl_xor(t8[r], off, 16));

    float p0a[8], p1a[8], rs[8], corr[8];
#pragma unroll
    for (int r = 0; r < 8; ++r) {
      const float nm = fmaxf(run_m[r], t8[r]);
      corr[r] = __expf(run_m[r] - nm);
      run_m[r] = nm;
      p0a[r] = __expf(s[0][r] - nm);
      p1a[r] = __expf(s[1][r] - nm);
      rs[r]  = p0a[r] + p1a[r];
    }
#pragma unroll
    for (int off = 1; off < 16; off <<= 1)
#pragma unroll
      for (int r = 0; r < 8; ++r) rs[r] += __shfl_xor(rs[r], off, 16);
#pragma unroll
    for (int r = 0; r < 8; ++r) run_l[r] = run_l[r] * corr[r] + rs[r];
#pragma unroll
    for (int t = 0; t < 4; ++t)
#pragma unroll
      for (int r = 0; r < 8; ++r) O[t][r] *= corr[r];

    // transpose P (C/D layout) -> A layout via LDS (wave-private tile, so a
    // dscnt wait is enough: per-wave DS ops execute in order)
#pragma unroll
    for (int r = 0; r < 8; ++r) {
      Pt[wid][r + 8 * half][n]      = (_Float16)p0a[r];
      Pt[wid][r + 8 * half][16 + n] = (_Float16)p1a[r];
    }
    mha_wait_ds0();
    union { v16h v; v8h h[2]; } ap;
    ap.h[0] = *(const v8h*)(&Pt[wid][n][half * 16]);
    ap.h[1] = *(const v8h*)(&Pt[wid][n][half * 16 + 8]);

    // O += P @ V  (B-operand from LDS-resident transposed V)
#pragma unroll
    for (int t = 0; t < 4; ++t) {
      v16h bv = *(const v16h*)(&Vsh[cur][t * 16 + n][half * 16]);
      O[t] = WMMA_F16(ap.v, bv, O[t]);
    }
    cur = nxt;
  }

  // normalize and store into concat [ROWS, H*DV]
  const int bi = bh / H;
  const int hh = bh % H;
#pragma unroll
  for (int r = 0; r < 8; ++r) {
    const int row   = q0 + r + 8 * half;
    const float inv = (run_l[r] > 0.f) ? (1.0f / run_l[r]) : 0.f;
    const size_t rb = (size_t)(bi * S + row) * FEAT + hh * DV;
#pragma unroll
    for (int t = 0; t < 4; ++t)
      concat[rb + t * 16 + n] = (_Float16)(O[t][r] * inv);
  }
}

// ---------------------------------------------------------------------------
// Kernel 3: output projection + bias + residual. 16x64 tile per wave.
// ---------------------------------------------------------------------------
__global__ __launch_bounds__(32)
void mha_out_gemm(const _Float16* __restrict__ ch, const _Float16* __restrict__ Woh,
                  const float* __restrict__ bo, const float* __restrict__ x,
                  float* __restrict__ zbuf) {
  const int lane = threadIdx.x;
  const int n    = lane & 15;
  const int half = lane >> 4;
  const int m0   = blockIdx.y * 16;
  const int n0   = blockIdx.x * 64;

  const v8f vzero = {0.f, 0.f, 0.f, 0.f, 0.f, 0.f, 0.f, 0.f};
  v8f acc[4] = {vzero, vzero, vzero, vzero};

  const _Float16* arow = ch + (size_t)(m0 + n) * FEAT;
  for (int k0 = 0; k0 < FEAT; k0 += 32) {
    __builtin_prefetch(arow + k0 + 64, 0, 0);
    union { v16h v; v8h h[2]; } a;
    a.h[0] = *(const v8h*)(arow + k0 + half * 16);
    a.h[1] = *(const v8h*)(arow + k0 + half * 16 + 8);
#pragma unroll
    for (int sub = 0; sub < 4; ++sub) {
      const _Float16* wrow = Woh + (size_t)(n0 + sub * 16 + n) * FEAT;
      v16h b = *(const v16h*)(wrow + k0 + half * 16);
      acc[sub] = WMMA_F16(a.v, b, acc[sub]);
    }
  }

#pragma unroll
  for (int sub = 0; sub < 4; ++sub) {
    const int col = n0 + sub * 16 + n;
    const float bias = bo[col];
#pragma unroll
    for (int r = 0; r < 8; ++r) {
      const int row = m0 + r + 8 * half;
      const size_t idx = (size_t)row * D + col;
      zbuf[idx] = acc[sub][r] + bias + x[idx];
    }
  }
}

// ---------------------------------------------------------------------------
// Kernel 4: layernorm over D=512 per row. 128 threads (4 waves) per row.
// ---------------------------------------------------------------------------
__global__ __launch_bounds__(128)
void mha_layernorm(const float* __restrict__ zbuf, const float* __restrict__ gamma,
                   const float* __restrict__ beta, float* __restrict__ out) {
  const int row = blockIdx.x;
  const int tid = threadIdx.x;
  const float4 v = ((const float4*)(zbuf + (size_t)row * D))[tid];

  float sum = v.x + v.y + v.z + v.w;
  float sq  = v.x * v.x + v.y * v.y + v.z * v.z + v.w * v.w;
#pragma unroll
  for (int off = 1; off < 32; off <<= 1) {
    sum += __shfl_xor(sum, off, 32);
    sq  += __shfl_xor(sq,  off, 32);
  }
  __shared__ float ssum[4], ssq[4];
  const int wid = tid >> 5;
  if ((tid & 31) == 0) { ssum[wid] = sum; ssq[wid] = sq; }
  __syncthreads();
  sum = ssum[0] + ssum[1] + ssum[2] + ssum[3];
  sq  = ssq[0]  + ssq[1]  + ssq[2]  + ssq[3];

  const float mu   = sum * (1.0f / D);
  const float var  = sq * (1.0f / D) - mu * mu;
  const float rstd = rsqrtf(var + 1e-5f);

  const float4 g  = ((const float4*)gamma)[tid];
  const float4 bb = ((const float4*)beta)[tid];
  float4 o;
  o.x = (v.x - mu) * rstd * g.x + bb.x;
  o.y = (v.y - mu) * rstd * g.y + bb.y;
  o.z = (v.z - mu) * rstd * g.z + bb.z;
  o.w = (v.w - mu) * rstd * g.w + bb.w;
  ((float4*)(out + (size_t)row * D))[tid] = o;
}

// ---------------------------------------------------------------------------
extern "C" void kernel_launch(void* const* d_in, const int* in_sizes, int n_in,
                              void* d_out, int out_size, void* d_ws, size_t ws_size,
                              hipStream_t stream) {
  (void)in_sizes; (void)n_in; (void)out_size; (void)ws_size;
  const float* x     = (const float*)d_in[0];
  const float* Wp    = (const float*)d_in[1];
  const float* bp    = (const float*)d_in[2];
  const float* Wo    = (const float*)d_in[3];
  const float* bo    = (const float*)d_in[4];
  const float* gamma = (const float*)d_in[5];
  const float* beta  = (const float*)d_in[6];
  float* out = (float*)d_out;

  char* ws = (char*)d_ws;
  _Float16* xh     = (_Float16*)(ws + OFF_XH);
  _Float16* Wph    = (_Float16*)(ws + OFF_WPH);
  _Float16* Woh    = (_Float16*)(ws + OFF_WOH);
  _Float16* qh     = (_Float16*)(ws + OFF_QH);
  _Float16* kh     = (_Float16*)(ws + OFF_KH);
  _Float16* vth    = (_Float16*)(ws + OFF_VTH);
  _Float16* concat = (_Float16*)(ws + OFF_CONCAT);
  float*    zbuf   = (float*)(ws + OFF_Z);

  mha_cvt_f16<<<2048, 256, 0, stream>>>(x,  xh,  ROWS * D);
  mha_cvt_f16<<<1024, 256, 0, stream>>>(Wp, Wph, PROJ * D);
  mha_cvt_f16<<<512,  256, 0, stream>>>(Wo, Woh, D * FEAT);

  mha_qkv_gemm<<<dim3(PROJ / 64, ROWS / 16), 32, 0, stream>>>(xh, Wph, bp, qh, kh, vth);
  mha_flash_attn<<<dim3(S / 64, BN * H), 128, 0, stream>>>(qh, kh, vth, concat);
  mha_out_gemm<<<dim3(D / 64, ROWS / 16), 32, 0, stream>>>(concat, Woh, bo, x, zbuf);
  mha_layernorm<<<ROWS, 128, 0, stream>>>(zbuf, gamma, beta, out);
}
